// SelfAttention_41893111005222
// MI455X (gfx1250) — matrix operand order repaired and checked by reading, not compiled, not run on hardware
//
#include <hip/hip_runtime.h>
#include <math.h>

typedef __attribute__((ext_vector_type(16))) _Float16 v16h;
typedef __attribute__((ext_vector_type(8)))  _Float16 v8h;
typedef __attribute__((ext_vector_type(2)))  _Float16 v2h;
typedef __attribute__((ext_vector_type(8)))  float    v8f;

typedef unsigned int u32;
typedef unsigned long long u64;
typedef __attribute__((ext_vector_type(4))) u32 u32x4;
typedef __attribute__((ext_vector_type(8))) int i32x8;
typedef __attribute__((ext_vector_type(4))) int i32x4;

#define B_  2
#define S_  2048
#define H_  16
#define D_  64
#define RS_ (3*H_*D_)   /* seq-stride in floats inside qkv: (B,S,3,H,D) */

#define WMMA(a,b,c) __builtin_amdgcn_wmma_f32_16x16x32_f16(false,(a),false,(b),(short)0,(c),false,false)

// Raw v_exp_f32 (2^x). Inputs are far from the denorm region; masked -1e30 -> 0.
__device__ __forceinline__ float fast_exp2(float x) { return __builtin_amdgcn_exp2f(x); }

// TDM: DMA one 32-row x 64-float f32 tile (row stride RS_) into LDS at lds_off.
// Used purely as a prefetcher (dest buffer is never read) -> correctness-neutral.
__device__ __forceinline__ void tdm_prefetch_tile(const float* gsrc, u32 lds_off) {
  u64 ga = (u64)(uintptr_t)gsrc;
  u32x4 g0;
  g0[0] = 1u;                                            // count=1 (valid), user mode
  g0[1] = lds_off;                                       // lds_addr (bytes)
  g0[2] = (u32)ga;                                       // global_addr[31:0]
  g0[3] = (u32)((ga >> 32) & 0x01FFFFFFu) | (2u << 30);  // global_addr[56:32] | type=2
  i32x8 g1;
  g1[0] = (int)(2u << 16);   // workgroup_mask=0, data_size=2 (4 bytes)
  g1[1] = (int)(64u << 16);  // tensor_dim0 = 64 (bits 79:48, low 16 here)
  g1[2] = (int)(32u << 16);  // tensor_dim0 hi=0 | tensor_dim1 = 32 (low 16)
  g1[3] = (int)(64u << 16);  // tensor_dim1 hi=0 | tile_dim0 = 64
  g1[4] = 32;                // tile_dim1 = 32, tile_dim2 = 0
  g1[5] = RS_;               // tensor_dim0_stride = 3072 (low 32 of 48)
  g1[6] = 0;                 // stride hi | tensor_dim1_stride lo (unused, 2D)
  g1[7] = 0;
  i32x4 z4 = {0, 0, 0, 0};
  i32x8 z8 = {0, 0, 0, 0, 0, 0, 0, 0};
  __builtin_amdgcn_tensor_load_to_lds(g0, g1, z4, z4, z8, 0);
}

// Assemble a 16-half fragment from 32 contiguous bytes of LDS (two b128 reads).
__device__ __forceinline__ v16h ldfrag_lds(const _Float16* p) {
  v8h a = *(const v8h*)p;
  v8h b = *(const v8h*)(p + 8);
  v16h r;
#pragma unroll
  for (int i = 0; i < 8; ++i) { r[i] = a[i]; r[i + 8] = b[i]; }
  return r;
}

// Load 8 consecutive f32 from global, scale, convert, place at r[at..at+7].
__device__ __forceinline__ void ld8s(const float* p, float s, v16h& r, int at) {
  const float4* q = (const float4*)p;
  float4 a = q[0], b = q[1];
  r[at+0]=(_Float16)(a.x*s); r[at+1]=(_Float16)(a.y*s);
  r[at+2]=(_Float16)(a.z*s); r[at+3]=(_Float16)(a.w*s);
  r[at+4]=(_Float16)(b.x*s); r[at+5]=(_Float16)(b.y*s);
  r[at+6]=(_Float16)(b.z*s); r[at+7]=(_Float16)(b.w*s);
}

// One 32-wide kv block: QK^T (4 WMMA), exp2 softmax numerator, P->LDS->A-frag,
// P*V (4 WMMA) + rowsum via P*ones (1 WMMA). MASKED only for the diagonal block.
template<bool MASKED>
__device__ __forceinline__ void attn_block(
    int kv0, int q_base, int half_id, int l16,
    const v16h& aq0, const v16h& aq1, const v16h& ones,
    const _Float16 (*kbuf)[64], const _Float16 (*vtbuf)[32], _Float16 (*pb)[32],
    v8f& acc0, v8f& acc1, v8f& acc2, v8f& acc3, v8f& lsum) {

  const int dbase = half_id ? 16 : 0;
  const _Float16* kr0 = &kbuf[l16][dbase];
  const _Float16* kr1 = &kbuf[16 + l16][dbase];
  v16h bk0l = ldfrag_lds(kr0);
  v16h bk0h = ldfrag_lds(kr0 + 32);
  v16h bk1l = ldfrag_lds(kr1);
  v16h bk1h = ldfrag_lds(kr1 + 32);

  v8f s0 = {}, s1 = {};
  s0 = WMMA(aq0, bk0l, s0);
  s0 = WMMA(aq1, bk0h, s0);
  s1 = WMMA(aq0, bk1l, s1);
  s1 = WMMA(aq1, bk1h, s1);

  // scores pre-scaled by log2(e)/sqrt(D): p = 2^s == e^(qk/sqrt(D)), exact.
#pragma unroll
  for (int r = 0; r < 8; ++r) {
    float x0 = s0[r], x1 = s1[r];
    if (MASKED) {
      const int grow = q_base + r + 8 * half_id;
      if (kv0 + l16      > grow) x0 = -1.0e30f;
      if (kv0 + 16 + l16 > grow) x1 = -1.0e30f;
    }
    const float p0 = fast_exp2(x0);
    const float p1 = fast_exp2(x1);
    const int mloc = r + 8 * half_id;
    pb[mloc][l16]      = (_Float16)p0;
    pb[mloc][l16 + 16] = (_Float16)p1;
  }
  // Per-wave LDS slice; same-wave DS ops are in-order -> no barrier needed.

  v16h ap;
  {
    const v8h* prow = (const v8h*)(&pb[l16][0]);
    v8h plo = prow[2 * half_id];
    v8h phi = prow[2 * half_id + 1];
#pragma unroll
    for (int i = 0; i < 8; ++i) { ap[i] = plo[i]; ap[i + 8] = phi[i]; }
  }

  const int koff = half_id ? 16 : 0;
  v16h bv0 = ldfrag_lds(&vtbuf[ 0 + l16][koff]);
  v16h bv1 = ldfrag_lds(&vtbuf[16 + l16][koff]);
  v16h bv2 = ldfrag_lds(&vtbuf[32 + l16][koff]);
  v16h bv3 = ldfrag_lds(&vtbuf[48 + l16][koff]);

  acc0 = WMMA(ap, bv0, acc0);
  acc1 = WMMA(ap, bv1, acc1);
  acc2 = WMMA(ap, bv2, acc2);
  acc3 = WMMA(ap, bv3, acc3);
  lsum = WMMA(ap, ones, lsum);   // row sums of P, broadcast across columns
}

__launch_bounds__(128)
__global__ void fa_causal_wmma(const float* __restrict__ qkv, float* __restrict__ out) {
  const int WGQT = S_ / 64;
  const int idx  = blockIdx.x;
  const int qtw  = idx % WGQT;
  const int h    = (idx / WGQT) % H_;
  const int b    = idx / (WGQT * H_);

  const int tid     = threadIdx.x;            // 0..127 (4 waves)
  const int wave    = tid >> 5;
  const int lane    = tid & 31;
  const int half_id = lane >> 4;
  const int l16     = lane & 15;

  const int q_base = qtw * 64 + wave * 16;

  const float* Qb = qkv + (size_t)b*S_*RS_ + 0*H_*D_ + h*D_;
  const float* Kb = qkv + (size_t)b*S_*RS_ + 1*H_*D_ + h*D_;
  const float* Vb = qkv + (size_t)b*S_*RS_ + 2*H_*D_ + h*D_;

  __shared__ __align__(16) _Float16 kbuf[32][64];     // [kv][d]
  __shared__ __align__(16) _Float16 vtbuf[64][32];    // [d][kv] (transposed)
  __shared__ __align__(16) _Float16 pbuf[4][16][32];  // per-wave P staging
  __shared__ __align__(16) float    pfbuf[2][32][64]; // TDM prefetch landing zone

  // Q fragments, scale = log2(e)/sqrt(D) folded in so p = exp2(score).
  const float scale = 0.125f * 1.44269504088896f;
  const float* qptr = Qb + (size_t)(q_base + l16) * RS_;
  const int kb = half_id ? 16 : 0;
  v16h aq0, aq1;
  ld8s(qptr + kb,           scale, aq0, 0);
  ld8s(qptr + kb + 8,       scale, aq0, 8);
  ld8s(qptr + 32 + kb,      scale, aq1, 0);
  ld8s(qptr + 32 + kb + 8,  scale, aq1, 8);

  v16h ones;
#pragma unroll
  for (int i = 0; i < 16; ++i) ones[i] = (_Float16)1.0f;

  v8f acc0 = {}, acc1 = {}, acc2 = {}, acc3 = {}, lsum = {};

  // Staging split for K: 1 kv row x 16 d per thread.
  const int srow   = tid >> 2;
  const int schunk = (tid & 3) * 16;
  // Staging split for V: 2 adjacent kv rows x 8 d per thread (b32 packed stores).
  const int vrow   = (tid >> 3) * 2;
  const int vchunk = (tid & 7) * 8;

  const u32 pf_k = (u32)(uintptr_t)&pfbuf[0][0][0];
  const u32 pf_v = (u32)(uintptr_t)&pfbuf[1][0][0];

  const int wb   = q_base >> 5;               // this wave's diagonal block
  const int nblk = qtw * 2 + 2;               // workgroup kv-block count

  for (int jb = 0; jb < nblk; ++jb) {
    const int kv0 = jb * 32;

    // ---------- TDM prefetch of next block's K/V tiles (data-flow neutral) ----------
    if (wave == 0 && jb + 1 < nblk) {
      __builtin_amdgcn_s_wait_tensorcnt(4);
      tdm_prefetch_tile(Kb + (size_t)(kv0 + 32) * RS_, pf_k);
      tdm_prefetch_tile(Vb + (size_t)(kv0 + 32) * RS_, pf_v);
    }

    // ---------- cooperative stage: K, V -> LDS (f16, V transposed) ----------
    {
      // K: row-major f16
      const float4* ks = (const float4*)(Kb + (size_t)(kv0 + srow) * RS_ + schunk);
      float4 k0 = ks[0], k1 = ks[1], k2 = ks[2], k3 = ks[3];
      v8h h0, h1;
      h0[0]=(_Float16)k0.x; h0[1]=(_Float16)k0.y; h0[2]=(_Float16)k0.z; h0[3]=(_Float16)k0.w;
      h0[4]=(_Float16)k1.x; h0[5]=(_Float16)k1.y; h0[6]=(_Float16)k1.z; h0[7]=(_Float16)k1.w;
      h1[0]=(_Float16)k2.x; h1[1]=(_Float16)k2.y; h1[2]=(_Float16)k2.z; h1[3]=(_Float16)k2.w;
      h1[4]=(_Float16)k3.x; h1[5]=(_Float16)k3.y; h1[6]=(_Float16)k3.z; h1[7]=(_Float16)k3.w;
      *(v8h*)&kbuf[srow][schunk]     = h0;
      *(v8h*)&kbuf[srow][schunk + 8] = h1;

      // V: transposed f16, two rows packed per b32 store
      const float4* va = (const float4*)(Vb + (size_t)(kv0 + vrow)     * RS_ + vchunk);
      const float4* vb = (const float4*)(Vb + (size_t)(kv0 + vrow + 1) * RS_ + vchunk);
      float4 a0 = va[0], a1 = va[1];
      float4 b0 = vb[0], b1 = vb[1];
      float fa[8] = { a0.x,a0.y,a0.z,a0.w, a1.x,a1.y,a1.z,a1.w };
      float fb[8] = { b0.x,b0.y,b0.z,b0.w, b1.x,b1.y,b1.z,b1.w };
#pragma unroll
      for (int i = 0; i < 8; ++i) {
        v2h t; t[0] = (_Float16)fa[i]; t[1] = (_Float16)fb[i];
        *(v2h*)&vtbuf[vchunk + i][vrow] = t;
      }
    }
    __syncthreads();

    // ---------- per-wave compute, wave-uniform predicates ----------
    if (jb < wb) {
      attn_block<false>(kv0, q_base, half_id, l16, aq0, aq1, ones,
                        kbuf, vtbuf, pbuf[wave], acc0, acc1, acc2, acc3, lsum);
    } else if (jb == wb) {
      attn_block<true>(kv0, q_base, half_id, l16, aq0, aq1, ones,
                       kbuf, vtbuf, pbuf[wave], acc0, acc1, acc2, acc3, lsum);
    }
    __syncthreads();   // all waves done reading kbuf/vtbuf before restage
  }

  // ---------- epilogue: normalize (v_rcp_f32), store fp32 ----------
#pragma unroll
  for (int r = 0; r < 8; ++r) {
    const float inv = __builtin_amdgcn_rcpf(lsum[r]);
    const int grow = q_base + r + 8 * half_id;
    float* orow = out + ((size_t)(b * S_ + grow) * H_ + h) * D_;
    orow[l16]      = acc0[r] * inv;
    orow[16 + l16] = acc1[r] * inv;
    orow[32 + l16] = acc2[r] * inv;
    orow[48 + l16] = acc3[r] * inv;
  }
}

extern "C" void kernel_launch(void* const* d_in, const int* in_sizes, int n_in,
                              void* d_out, int out_size, void* d_ws, size_t ws_size,
                              hipStream_t stream) {
  (void)in_sizes; (void)n_in; (void)d_ws; (void)ws_size; (void)out_size;
  const float* qkv = (const float*)d_in[0];
  float* out = (float*)d_out;
  dim3 grid(B_ * H_ * (S_ / 64));
  fa_causal_wmma<<<grid, 128, 0, stream>>>(qkv, out);
}
